// PNAConvHetero_46574625358330
// MI455X (gfx1250) — compile-verified, matrix operand order checked by reading, not run on hardware
//
#include <hip/hip_runtime.h>
#include <math.h>

// PNAConvHetero for MI455X (gfx1250, wave32, WMMA f32<=f16xf16).
// Pipeline (all on `stream`, workspace-only scratch, deterministic work graph):
//   1) degrees (atomicAdd) + avg_log reduction
//   2) weights fp32 -> f16 pre-shuffled into WMMA B-fragment layout
//   3) per direction: edge kernel (edge-encoder WMMA + pre_nn WMMA + 4-way
//      scatter atomics), node kernel (aggregate finalize + post_nn/lin WMMA)
//   4) final concat GEMM -> d_out
// Workspace need ~155 MB (6x N*64 f32 buffers + degrees + f16 weight frags).

typedef __attribute__((ext_vector_type(16))) _Float16 v16h;
typedef __attribute__((ext_vector_type(4)))  _Float16 v4h;
typedef __attribute__((ext_vector_type(8)))  float    v8f;

#define PNA_EPS 1e-5f

__device__ __forceinline__ v8f wmma_f16(v16h a, v16h b, v8f c) {
  return __builtin_amdgcn_wmma_f32_16x16x32_f16(false, a, false, b, (short)0, c,
                                                false, false);
}

// 16-bit A-matrix 16x32 layout (ISA 7.12.2): lane<16 holds K 0-7,16-23;
// lane>=16 holds K 8-15,24-31 for row M = lane&15. B uses the mirrored layout
// with columns per lane; B fragments are pre-shuffled so each lane loads 16
// contiguous halves (two global_load_b128).
__device__ __forceinline__ v16h load_a_frag_lds(const _Float16* base, int stride,
                                                int lane, int kbase) {
  const _Float16* p = base + (lane & 15) * stride + kbase + ((lane & 16) >> 1);
  v16h a;
#pragma unroll
  for (int h = 0; h < 16; ++h) a[h] = p[(h & 7) + ((h & 8) << 1)];
  return a;
}

__device__ __forceinline__ v16h load_b_frag(const _Float16* F, int kc, int nb,
                                            int lane) {
  const v16h* p = (const v16h*)F;
  return p[(((kc << 2) + nb) << 5) + lane];
}

__device__ __forceinline__ v4h f4_to_h4(float4 v) {
  v4h h;
  h[0] = (_Float16)v.x; h[1] = (_Float16)v.y;
  h[2] = (_Float16)v.z; h[3] = (_Float16)v.w;
  return h;
}

__device__ __forceinline__ void atomicMinF(float* a, float v) {
  if (v >= 0.0f) atomicMin((int*)a, __float_as_int(v));
  else           atomicMax((unsigned int*)a, __float_as_uint(v));
}
__device__ __forceinline__ void atomicMaxF(float* a, float v) {
  if (v >= 0.0f) atomicMax((int*)a, __float_as_int(v));
  else           atomicMin((unsigned int*)a, __float_as_uint(v));
}

// ---------------------------------------------------------------- small kernels

__global__ void k_init_deg(float* degD, float* degS, float* avgb, int N) {
  int i = blockIdx.x * blockDim.x + threadIdx.x;
  if (i < N) { degD[i] = 0.0f; degS[i] = 0.0f; }
  if (i < 2) avgb[i] = 0.0f;
}

__global__ void k_deg(const int* __restrict__ src, const int* __restrict__ dst,
                      float* degD, float* degS, int E) {
  int i = blockIdx.x * blockDim.x + threadIdx.x;
  if (i < E) {
    atomicAdd(&degD[dst[i]], 1.0f);
    atomicAdd(&degS[src[i]], 1.0f);
  }
}

__global__ void k_avg_reduce(const float* __restrict__ deg, float* avgb, int N) {
  __shared__ float red[256];
  float s = 0.0f;
  for (int i = blockIdx.x * blockDim.x + threadIdx.x; i < N;
       i += gridDim.x * blockDim.x)
    s += logf(deg[i] + 1.0f);
  red[threadIdx.x] = s;
  __syncthreads();
  for (int o = 128; o > 0; o >>= 1) {
    if ((int)threadIdx.x < o) red[threadIdx.x] += red[threadIdx.x + o];
    __syncthreads();
  }
  if (threadIdx.x == 0) atomicAdd(&avgb[0], red[0]);
}

__global__ void k_avg_final(float* avgb, int N) {
  if (blockIdx.x == 0 && threadIdx.x == 0) avgb[1] = avgb[0] / (float)N;
}

// Convert fp32 W[K][64] into per-lane WMMA B fragments (f16), 16 halves/lane.
__global__ void k_wfrag(const float* __restrict__ W, _Float16* __restrict__ F,
                        int K) {
  int idx = blockIdx.x * blockDim.x + threadIdx.x;
  if (idx >= K * 64) return;
  int h    = idx & 15;
  int lane = (idx >> 4) & 31;
  int t    = idx >> 9;      // kc*4 + nb
  int nb   = t & 3;
  int kc   = t >> 2;
  int k    = kc * 32 + (h & 7) + ((h & 8) << 1) + ((lane & 16) >> 1);
  int col  = nb * 16 + (lane & 15);
  F[idx] = (_Float16)W[k * 64 + col];
}

__global__ void k_init_agg(float* s, float* q, float* mn, float* mx, int n) {
  int i = blockIdx.x * blockDim.x + threadIdx.x;
  if (i < n) {
    s[i] = 0.0f; q[i] = 0.0f;
    mn[i] = __builtin_inff();
    mx[i] = -__builtin_inff();
  }
}

// ---------------------------------------------------------------- edge kernel
// 8 waves/block, 16 edges per wave. Per wave:
//   e = eattr@We + be  (1 kc x 4 nb WMMA), staged as A cols 128..191
//   h = [x_recv | x_send | e] @ Wpre + bpre  (6 kc x 4 nb WMMA)
//   scatter sum/sumsq/min/max onto recv nodes (L2-resident atomics)
#define EW 8
__global__ __launch_bounds__(256) void k_edge(
    const float* __restrict__ x, const int* __restrict__ recvIdx,
    const int* __restrict__ sendIdx, const float* __restrict__ eattr,
    const _Float16* __restrict__ WeF, const float* __restrict__ be,
    const _Float16* __restrict__ WpreF, const float* __restrict__ bpre,
    float* __restrict__ sumB, float* __restrict__ sqB, float* __restrict__ mnB,
    float* __restrict__ mxB, int E) {
  __shared__ _Float16 A[EW][16][192];
  __shared__ _Float16 Ea[EW][16][32];
  __shared__ int nR[EW][16];
  __shared__ int nS[EW][16];
  __shared__ int vld[EW][16];

  const int lane  = threadIdx.x & 31;
  const int w     = threadIdx.x >> 5;
  const int ebase = (blockIdx.x * EW + w) * 16;

  if (lane < 16) {
    int e  = ebase + lane;
    int ok = (e < E) ? 1 : 0;
    int ec = ok ? e : 0;
    nR[w][lane]  = recvIdx[ec];
    nS[w][lane]  = sendIdx[ec];
    vld[w][lane] = ok;
  }
  // prefetch next block's edge_attr tile (wave-uniform guard; 64B/lane = 2KB)
  {
    int ne = ebase + EW * 16;
    if (ne < E) __builtin_prefetch(eattr + (size_t)ne * 32 + lane * 16, 0, 0);
  }
  __syncthreads();

  // stage x rows (float4 gathers -> packed f16x4 LDS stores)
  for (int i = lane; i < 16 * 16; i += 32) {
    int r = i >> 4, q = i & 15;
    float4 vr = ((const float4*)(x + (size_t)nR[w][r] * 64))[q];
    float4 vs = ((const float4*)(x + (size_t)nS[w][r] * 64))[q];
    *(v4h*)&A[w][r][q * 4]      = f4_to_h4(vr);
    *(v4h*)&A[w][r][64 + q * 4] = f4_to_h4(vs);
  }
  for (int i = lane; i < 16 * 8; i += 32) {
    int r = i >> 3, q = i & 7;
    int e = ebase + r;
    if (e >= E) e = 0;
    float4 v = ((const float4*)(eattr + (size_t)e * 32))[q];
    *(v4h*)&Ea[w][r][q * 4] = f4_to_h4(v);
  }
  __syncthreads();

  {  // edge encoder GEMM [16x32] x [32x64]
    v16h a = load_a_frag_lds(&Ea[w][0][0], 32, lane, 0);
#pragma unroll
    for (int nb = 0; nb < 4; ++nb) {
      v8f c = {};
      c = wmma_f16(a, load_b_frag(WeF, 0, nb, lane), c);
      int col  = (lane & 15) + nb * 16;
      float bb = be[col];
#pragma unroll
      for (int v = 0; v < 8; ++v) {
        int r = v + ((lane >> 4) << 3);
        A[w][r][128 + col] = (_Float16)(c[v] + bb);
      }
    }
  }
  __syncthreads();

  v8f acc[4] = {};
#pragma unroll
  for (int kc = 0; kc < 6; ++kc) {
    v16h a = load_a_frag_lds(&A[w][0][0], 192, lane, kc * 32);
#pragma unroll
    for (int nb = 0; nb < 4; ++nb)
      acc[nb] = wmma_f16(a, load_b_frag(WpreF, kc, nb, lane), acc[nb]);
  }

#pragma unroll
  for (int nb = 0; nb < 4; ++nb) {
    int col  = (lane & 15) + nb * 16;
    float bb = bpre[col];
#pragma unroll
    for (int v = 0; v < 8; ++v) {
      int r = v + ((lane >> 4) << 3);
      if (vld[w][r]) {
        float h  = acc[nb][v] + bb;
        size_t o = (size_t)nR[w][r] * 64 + col;
        atomicAdd(&sumB[o], h);
        atomicAdd(&sqB[o], h * h);
        atomicMinF(&mnB[o], h);
        atomicMaxF(&mxB[o], h);
      }
    }
  }
}

// ---------------------------------------------------------------- node kernel
// [x|agg|agg*s1|agg*s2]@Wpost == x@W0 + agg@W1 + s1*(agg@W2) + s2*(agg@W3);
// then @Wlin. 4 waves/block, 16 nodes per wave.
#define NW 4
__global__ __launch_bounds__(128) void k_node(
    const float* __restrict__ x, const float* __restrict__ deg,
    const float* __restrict__ avgb, const float* __restrict__ sumB,
    const float* __restrict__ sqB, const float* __restrict__ mnB,
    const float* __restrict__ mxB, const _Float16* __restrict__ W0F,
    const _Float16* __restrict__ W1F, const _Float16* __restrict__ W2F,
    const _Float16* __restrict__ W3F, const float* __restrict__ bpost,
    const _Float16* __restrict__ WlinF, const float* __restrict__ blin,
    float* __restrict__ outA, int N) {
  __shared__ _Float16 AG[NW][16][256];
  __shared__ _Float16 X[NW][16][64];
  __shared__ _Float16 H[NW][16][64];
  __shared__ float s1s[NW][16], s2s[NW][16], c1s[NW][16];
  __shared__ int nid[NW][16], vldn[NW][16], nonz[NW][16];

  const int lane  = threadIdx.x & 31;
  const int w     = threadIdx.x >> 5;
  const int nbase = (blockIdx.x * NW + w) * 16;
  const float avg = avgb[1];

  if (lane < 16) {
    int n      = nbase + lane;
    int ok     = (n < N) ? 1 : 0;
    int nc     = ok ? n : 0;
    float cnt  = deg[nc];
    float c1   = fmaxf(cnt, 1.0f);
    float logd = logf(c1 + 1.0f);
    c1s[w][lane]  = c1;
    s1s[w][lane]  = logd / avg;
    s2s[w][lane]  = avg / logd;
    nid[w][lane]  = nc;
    vldn[w][lane] = ok;
    nonz[w][lane] = (cnt > 0.0f) ? 1 : 0;
  }
  __syncthreads();

  // finalize aggregators + stage rows (float4 loads, packed f16x4 LDS stores)
  for (int i = lane; i < 16 * 16; i += 32) {
    int r = i >> 4, q = i & 15;
    size_t o  = (size_t)nid[w][r] * 16 + q;  // float4 units
    float4 s4 = ((const float4*)sumB)[o];
    float4 q4 = ((const float4*)sqB)[o];
    float4 n4 = ((const float4*)mnB)[o];
    float4 m4 = ((const float4*)mxB)[o];
    float4 x4 = ((const float4*)x)[o];
    float rc  = 1.0f / c1s[w][r];
    int nz    = nonz[w][r];
    float4 mean, sd, mnv, mxv;
    mean.x = s4.x * rc; mean.y = s4.y * rc;
    mean.z = s4.z * rc; mean.w = s4.w * rc;
    sd.x = sqrtf(fmaxf(q4.x * rc - mean.x * mean.x, 0.0f) + PNA_EPS);
    sd.y = sqrtf(fmaxf(q4.y * rc - mean.y * mean.y, 0.0f) + PNA_EPS);
    sd.z = sqrtf(fmaxf(q4.z * rc - mean.z * mean.z, 0.0f) + PNA_EPS);
    sd.w = sqrtf(fmaxf(q4.w * rc - mean.w * mean.w, 0.0f) + PNA_EPS);
    mnv.x = nz ? n4.x : 0.0f; mnv.y = nz ? n4.y : 0.0f;
    mnv.z = nz ? n4.z : 0.0f; mnv.w = nz ? n4.w : 0.0f;
    mxv.x = nz ? m4.x : 0.0f; mxv.y = nz ? m4.y : 0.0f;
    mxv.z = nz ? m4.z : 0.0f; mxv.w = nz ? m4.w : 0.0f;
    *(v4h*)&AG[w][r][q * 4]       = f4_to_h4(mean);
    *(v4h*)&AG[w][r][64 + q * 4]  = f4_to_h4(mnv);
    *(v4h*)&AG[w][r][128 + q * 4] = f4_to_h4(mxv);
    *(v4h*)&AG[w][r][192 + q * 4] = f4_to_h4(sd);
    *(v4h*)&X[w][r][q * 4]        = f4_to_h4(x4);
  }
  __syncthreads();

#pragma unroll
  for (int nb = 0; nb < 4; ++nb) {
    v8f ax = {};
#pragma unroll
    for (int kc = 0; kc < 2; ++kc) {
      v16h a = load_a_frag_lds(&X[w][0][0], 64, lane, kc * 32);
      ax = wmma_f16(a, load_b_frag(W0F, kc, nb, lane), ax);
    }
    v8f a1 = {}, a2 = {}, a3 = {};
#pragma unroll
    for (int kc = 0; kc < 8; ++kc) {
      v16h a = load_a_frag_lds(&AG[w][0][0], 256, lane, kc * 32);
      a1 = wmma_f16(a, load_b_frag(W1F, kc, nb, lane), a1);
      a2 = wmma_f16(a, load_b_frag(W2F, kc, nb, lane), a2);
      a3 = wmma_f16(a, load_b_frag(W3F, kc, nb, lane), a3);
    }
    int col  = (lane & 15) + nb * 16;
    float bb = bpost[col];
#pragma unroll
    for (int v = 0; v < 8; ++v) {
      int r = v + ((lane >> 4) << 3);
      float o =
          ax[v] + a1[v] + s1s[w][r] * a2[v] + s2s[w][r] * a3[v] + bb;
      H[w][r][col] = (_Float16)o;
    }
  }
  __syncthreads();

#pragma unroll
  for (int nb = 0; nb < 4; ++nb) {
    v8f al = {};
#pragma unroll
    for (int kc = 0; kc < 2; ++kc) {
      v16h a = load_a_frag_lds(&H[w][0][0], 64, lane, kc * 32);
      al = wmma_f16(a, load_b_frag(WlinF, kc, nb, lane), al);
    }
    int col  = (lane & 15) + nb * 16;
    float bl = blin[col];
#pragma unroll
    for (int v = 0; v < 8; ++v) {
      int r = v + ((lane >> 4) << 3);
      if (vldn[w][r]) outA[(size_t)nid[w][r] * 64 + col] = al[v] + bl;
    }
  }
}

// ---------------------------------------------------------------- final kernel
#define FW 8
__global__ __launch_bounds__(256) void k_final(
    const float* __restrict__ x, const float* __restrict__ aIn,
    const float* __restrict__ aOut, const _Float16* __restrict__ WoutF,
    const float* __restrict__ bout, float* __restrict__ out, int N) {
  __shared__ _Float16 A[FW][16][192];
  __shared__ int nid[FW][16], vldn[FW][16];
  const int lane  = threadIdx.x & 31;
  const int w     = threadIdx.x >> 5;
  const int nbase = (blockIdx.x * FW + w) * 16;
  if (lane < 16) {
    int n = nbase + lane;
    int ok = (n < N) ? 1 : 0;
    nid[w][lane]  = ok ? n : 0;
    vldn[w][lane] = ok;
  }
  __syncthreads();
  for (int i = lane; i < 16 * 16; i += 32) {
    int r = i >> 4, q = i & 15;
    size_t o = (size_t)nid[w][r] * 16 + q;  // float4 units
    *(v4h*)&A[w][r][q * 4]       = f4_to_h4(((const float4*)x)[o]);
    *(v4h*)&A[w][r][64 + q * 4]  = f4_to_h4(((const float4*)aIn)[o]);
    *(v4h*)&A[w][r][128 + q * 4] = f4_to_h4(((const float4*)aOut)[o]);
  }
  __syncthreads();

  v8f acc[4] = {};
#pragma unroll
  for (int kc = 0; kc < 6; ++kc) {
    v16h a = load_a_frag_lds(&A[w][0][0], 192, lane, kc * 32);
#pragma unroll
    for (int nb = 0; nb < 4; ++nb)
      acc[nb] = wmma_f16(a, load_b_frag(WoutF, kc, nb, lane), acc[nb]);
  }
#pragma unroll
  for (int nb = 0; nb < 4; ++nb) {
    int col  = (lane & 15) + nb * 16;
    float bb = bout[col];
#pragma unroll
    for (int v = 0; v < 8; ++v) {
      int r = v + ((lane >> 4) << 3);
      if (vldn[w][r]) out[(size_t)nid[w][r] * 64 + col] = acc[nb][v] + bb;
    }
  }
}

// ---------------------------------------------------------------- launch

extern "C" void kernel_launch(void* const* d_in, const int* in_sizes, int n_in,
                              void* d_out, int out_size, void* d_ws,
                              size_t ws_size, hipStream_t stream) {
  const float* x       = (const float*)d_in[0];
  const int*   eidx    = (const int*)d_in[1];
  const float* eattr   = (const float*)d_in[2];
  const float* We_f    = (const float*)d_in[3];
  const float* Wpre_f  = (const float*)d_in[4];
  const float* Wpost_f = (const float*)d_in[5];
  const float* Wlin_f  = (const float*)d_in[6];
  const float* be_f    = (const float*)d_in[7];
  const float* bpre_f  = (const float*)d_in[8];
  const float* bpost_f = (const float*)d_in[9];
  const float* blin_f  = (const float*)d_in[10];
  const float* We_b    = (const float*)d_in[11];
  const float* Wpre_b  = (const float*)d_in[12];
  const float* Wpost_b = (const float*)d_in[13];
  const float* Wlin_b  = (const float*)d_in[14];
  const float* be_b    = (const float*)d_in[15];
  const float* bpre_b  = (const float*)d_in[16];
  const float* bpost_b = (const float*)d_in[17];
  const float* blin_b  = (const float*)d_in[18];
  const float* Wout    = (const float*)d_in[19];
  const float* bout    = (const float*)d_in[20];

  const int N = in_sizes[0] / 64;
  const int E = in_sizes[1] / 2;
  const int* srcI = eidx;
  const int* dstI = eidx + E;

  char* ws = (char*)d_ws;
  size_t off = 0;
  auto take = [&](size_t bytes) -> char* {
    char* p = ws + off;
    off = (off + bytes + 255) & ~(size_t)255;
    return p;
  };
  float* degD = (float*)take((size_t)N * 4);
  float* degS = (float*)take((size_t)N * 4);
  float* avgb = (float*)take(8);
  float* sumB = (float*)take((size_t)N * 64 * 4);
  float* sqB  = (float*)take((size_t)N * 64 * 4);
  float* mnB  = (float*)take((size_t)N * 64 * 4);
  float* mxB  = (float*)take((size_t)N * 64 * 4);
  float* aIn  = (float*)take((size_t)N * 64 * 4);
  float* aOut = (float*)take((size_t)N * 64 * 4);
  _Float16* weFf   = (_Float16*)take(32 * 64 * 2);
  _Float16* wpreFf = (_Float16*)take(192 * 64 * 2);
  _Float16* w0Ff   = (_Float16*)take(64 * 64 * 2);
  _Float16* w1Ff   = (_Float16*)take(256 * 64 * 2);
  _Float16* w2Ff   = (_Float16*)take(256 * 64 * 2);
  _Float16* w3Ff   = (_Float16*)take(256 * 64 * 2);
  _Float16* wlinFf = (_Float16*)take(64 * 64 * 2);
  _Float16* weFb   = (_Float16*)take(32 * 64 * 2);
  _Float16* wpreFb = (_Float16*)take(192 * 64 * 2);
  _Float16* w0Fb   = (_Float16*)take(64 * 64 * 2);
  _Float16* w1Fb   = (_Float16*)take(256 * 64 * 2);
  _Float16* w2Fb   = (_Float16*)take(256 * 64 * 2);
  _Float16* w3Fb   = (_Float16*)take(256 * 64 * 2);
  _Float16* wlinFb = (_Float16*)take(64 * 64 * 2);
  _Float16* woutF  = (_Float16*)take(192 * 64 * 2);

  // degrees + avg(log(deg_dst+1))
  k_init_deg<<<(N + 255) / 256, 256, 0, stream>>>(degD, degS, avgb, N);
  k_deg<<<(E + 255) / 256, 256, 0, stream>>>(srcI, dstI, degD, degS, E);
  k_avg_reduce<<<256, 256, 0, stream>>>(degD, avgb, N);
  k_avg_final<<<1, 1, 0, stream>>>(avgb, N);

  auto convW = [&](const float* W, _Float16* F, int K) {
    int n = K * 64;
    k_wfrag<<<(n + 255) / 256, 256, 0, stream>>>(W, F, K);
  };
  convW(We_f, weFf, 32);
  convW(Wpre_f, wpreFf, 192);
  convW(Wpost_f, w0Ff, 64);
  convW(Wpost_f + 64 * 64, w1Ff, 256);
  convW(Wpost_f + 320 * 64, w2Ff, 256);
  convW(Wpost_f + 576 * 64, w3Ff, 256);
  convW(Wlin_f, wlinFf, 64);
  convW(We_b, weFb, 32);
  convW(Wpre_b, wpreFb, 192);
  convW(Wpost_b, w0Fb, 64);
  convW(Wpost_b + 64 * 64, w1Fb, 256);
  convW(Wpost_b + 320 * 64, w2Fb, 256);
  convW(Wpost_b + 576 * 64, w3Fb, 256);
  convW(Wlin_b, wlinFb, 64);
  convW(Wout, woutF, 192);

  const int etiles  = (E + 15) / 16;
  const int eblocks = (etiles + EW - 1) / EW;
  const int ntiles  = (N + 15) / 16;
  const int nblocks = (ntiles + NW - 1) / NW;
  const int fblocks = (ntiles + FW - 1) / FW;
  const int aggn    = N * 64;

  // forward direction: messages into dst
  k_init_agg<<<(aggn + 255) / 256, 256, 0, stream>>>(sumB, sqB, mnB, mxB, aggn);
  k_edge<<<eblocks, 256, 0, stream>>>(x, dstI, srcI, eattr, weFf, be_f, wpreFf,
                                      bpre_f, sumB, sqB, mnB, mxB, E);
  k_node<<<nblocks, 128, 0, stream>>>(x, degD, avgb, sumB, sqB, mnB, mxB, w0Ff,
                                      w1Ff, w2Ff, w3Ff, bpost_f, wlinFf, blin_f,
                                      aIn, N);
  // backward direction: messages into src
  k_init_agg<<<(aggn + 255) / 256, 256, 0, stream>>>(sumB, sqB, mnB, mxB, aggn);
  k_edge<<<eblocks, 256, 0, stream>>>(x, srcI, dstI, eattr, weFb, be_b, wpreFb,
                                      bpre_b, sumB, sqB, mnB, mxB, E);
  k_node<<<nblocks, 128, 0, stream>>>(x, degS, avgb, sumB, sqB, mnB, mxB, w0Fb,
                                      w1Fb, w2Fb, w3Fb, bpost_b, wlinFb, blin_b,
                                      aOut, N);

  k_final<<<fblocks, 256, 0, stream>>>(x, aIn, aOut, woutF, bout, (float*)d_out,
                                       N);
}
